// MultiInputLSTMCell_15685220565615
// MI455X (gfx1250) — matrix operand order repaired and checked by reading, not compiled, not run on hardware
//
#include <hip/hip_runtime.h>
#include <hip/hip_bf16.h>

// ---------------------------------------------------------------------------
// MultiInputLSTMCell for MI455X (gfx1250, wave32, WMMA).
//
// Sizes: INPUT=1024, H=1024, C=32768.
// Dominant op: alpha_wh = c_input[32768x1024] @ alpha_weight_hh[1024x1024]
//   -> v_wmma_f32_16x16x32_bf16, fp32 accumulate, fused with the
//      sigmoid/exp/softmax-weighted child merge so c_input streams ~once.
//
// Workspace layout (floats at d_ws):
//   [0    ,1024) : aw      = input_ @ alpha_weight_ih + alpha_bias
//   [1024 ,2048) : i_raw   (gate pre-activations)
//   [2048 ,3072) : o_raw
//   [3072 ,4096) : g_raw
//   [4096 ,5120) : S = sum_c exp(sigmoid(alpha[c,h]))
//   [5120 ,6144) : P = sum_c exp(sigmoid(alpha[c,h])) * c_input[c,h]
//   byte 24576.. : BT = bf16 transpose of alpha_weight_hh, BT[n*1024+k] (2 MB)
// Requires ws_size >= 6144*4 + 2*1024*1024 bytes (~2.12 MB).
// ---------------------------------------------------------------------------

#define HID 1024
#define CNUM 32768

typedef __attribute__((ext_vector_type(16))) __bf16 bf16x16;
typedef __attribute__((ext_vector_type(8)))  __bf16 bf16x8;
typedef __attribute__((ext_vector_type(8)))  float  f32x8;

// ---------------------------------------------------------------- zero scratch
__global__ __launch_bounds__(256) void zero_kernel(float* __restrict__ wsf, int n) {
    int t = blockIdx.x * 256 + threadIdx.x;
    if (t < n) wsf[t] = 0.0f;
}

// ------------------------------------------------------- small GEMVs (gates)
// 16384 threads: 4 k-slices per output, 4096 outputs.
// outputs 0..3071   -> gates raw = bias + h0@Whh + x@Wih  (stored at wsf+1024)
// outputs 3072..4095 -> aw = abias + x@AWih               (stored at wsf+0)
__global__ __launch_bounds__(256) void gates_kernel(
    const float* __restrict__ inp, const float* __restrict__ h0,
    const float* __restrict__ wih, const float* __restrict__ whh,
    const float* __restrict__ awih,
    const float* __restrict__ bias, const float* __restrict__ abias,
    float* __restrict__ wsf)
{
    int t   = blockIdx.x * 256 + threadIdx.x;   // 0..16383
    int out = t >> 2;
    int ks  = (t & 3) * 256;
    float acc = 0.0f;
    if (out < 3 * HID) {
        for (int k = ks; k < ks + 256; ++k) {
            float w1 = whh[(size_t)k * (3 * HID) + out];
            float w2 = wih[(size_t)k * (3 * HID) + out];
            acc += h0[k] * w1 + inp[k] * w2;
        }
        if ((t & 3) == 0) acc += bias[out];
        atomicAdd(&wsf[1024 + out], acc);
    } else {
        int h = out - 3 * HID;
        for (int k = ks; k < ks + 256; ++k)
            acc += inp[k] * awih[(size_t)k * HID + h];
        if ((t & 3) == 0) acc += abias[h];
        atomicAdd(&wsf[h], acc);
    }
}

// --------------------------------------- transpose+convert alpha_weight_hh
// BT[n*1024+k] = (bf16) W[k*1024+n]; coalesced writes, 2 MB (L2-resident).
__global__ __launch_bounds__(256) void bt_kernel(
    const float* __restrict__ awhh, __bf16* __restrict__ BT)
{
    int t = blockIdx.x * 256 + threadIdx.x;     // 0..(1M-1)
    int k = t & (HID - 1);
    int n = t >> 10;
    BT[t] = (__bf16)awhh[(size_t)k * HID + n];
}

// --------------------------------------------------------------- main kernel
// Block: 256 threads = 8 waves; covers 16 children x full H (=1024).
// Wave w: 16 children x 128 h (8 WMMA tiles, 8x8=64 f32 accumulator VGPRs).
// Per k-step: issue 4 A loads, then all 16 B loads (one batch), convert A
// (only needs the in-order-completing A loads), then drain 8 WMMAs with
// staggered partial loadcnt waits. Fully unrolled -> no dynamic VGPR
// indexing -> no scratch spills.
__global__ __launch_bounds__(256) void lstm_child_kernel(
    const float*  __restrict__ c_input,
    const __bf16* __restrict__ BT,
    const float*  __restrict__ aw,
    float* __restrict__ S, float* __restrict__ P)
{
    const int lane    = threadIdx.x & 31;
    const int wave    = threadIdx.x >> 5;
    const int cbase   = blockIdx.x * 16;
    const int hbase   = wave * 128;
    const int halfsel = lane >> 4;      // 0 | 1
    const int ln      = lane & 15;

    f32x8 acc[8];
    #pragma unroll
    for (int j = 0; j < 8; ++j) acc[j] = {};

    // A: 16-bit 16x32 layout: lane holds row M=ln; elements 0..7 = K kb..kb+7,
    //    elements 8..15 = K kb+16..kb+23, kb = halfsel*8.
    const float* ap = c_input + (size_t)(cbase + ln) * HID;
    // B: 32x16, lane holds col N=ln; elements 0..7 = K kb..kb+7,
    //    elements 8..15 = K kb+16..kb+23, kb = halfsel*8 (same as A).
    const __bf16* bpbase = BT + (size_t)(hbase + ln) * HID + halfsel * 8;

    #pragma unroll 1
    for (int k0 = 0; k0 < HID; k0 += 32) {
        // ---- A loads first (complete in-order before the B batch) ----
        const float* apk = ap + k0 + halfsel * 8;
        float4 a0 = *(const float4*)(apk);
        float4 a1 = *(const float4*)(apk + 4);
        float4 a2 = *(const float4*)(apk + 16);
        float4 a3 = *(const float4*)(apk + 20);

        // ---- all 8 B fragments (16 x global_load_b128 batch) ----
        bf16x16 b[8];
        #pragma unroll
        for (int j = 0; j < 8; ++j) {
            const __bf16* bp = bpbase + (size_t)j * 16 * HID + k0;
            bf16x8 b0 = *(const bf16x8*)(bp);
            bf16x8 b1 = *(const bf16x8*)(bp + 16);
            #pragma unroll
            for (int e = 0; e < 8; ++e) { b[j][e] = b0[e]; b[j][8 + e] = b1[e]; }
        }

        // ---- convert A to bf16 (depends only on A loads) ----
        bf16x16 a;
        a[0]  = (__bf16)a0.x; a[1]  = (__bf16)a0.y; a[2]  = (__bf16)a0.z; a[3]  = (__bf16)a0.w;
        a[4]  = (__bf16)a1.x; a[5]  = (__bf16)a1.y; a[6]  = (__bf16)a1.z; a[7]  = (__bf16)a1.w;
        a[8]  = (__bf16)a2.x; a[9]  = (__bf16)a2.y; a[10] = (__bf16)a2.z; a[11] = (__bf16)a2.w;
        a[12] = (__bf16)a3.x; a[13] = (__bf16)a3.y; a[14] = (__bf16)a3.z; a[15] = (__bf16)a3.w;

        // ---- 8 independent WMMAs reusing the A fragment ----
        #pragma unroll
        for (int j = 0; j < 8; ++j) {
            acc[j] = __builtin_amdgcn_wmma_f32_16x16x32_bf16(
                false, a, false, b[j], (short)0, acc[j], false, false);
        }
    }

    // Epilogue (after ALL WMMAs; divergence allowed now). FULLY unrolled so
    // acc[j][r] stays in VGPRs (no dynamic indexing -> no scratch spill).
    // D layout: VGPR r, lanes 0-15 -> M=r, N=lane; lanes 16-31 -> M=8+r, N=lane-16.
    const int rowoff = cbase + halfsel * 8;
    #pragma unroll
    for (int j = 0; j < 8; ++j) {
        int   h    = hbase + j * 16 + ln;
        float awv  = aw[h];
        float esum = 0.0f, psum = 0.0f;
        #pragma unroll
        for (int r = 0; r < 8; ++r) {
            float x  = acc[j][r] + awv;
            float sg = 1.0f / (1.0f + __expf(-x));   // sigmoid(alpha)
            float e  = __expf(sg);                    // softmax numerator
            float cv = c_input[(size_t)(rowoff + r) * HID + h];
            esum += e;
            psum += e * cv;
        }
        esum += __shfl_xor(esum, 16, 32);
        psum += __shfl_xor(psum, 16, 32);
        if (lane < 16) {
            atomicAdd(&S[h], esum);
            atomicAdd(&P[h], psum);
        }
    }
}

// ----------------------------------------------------------------- finalize
__global__ __launch_bounds__(256) void finalize_kernel(
    const float* __restrict__ wsf, float* __restrict__ out)
{
    int h = blockIdx.x * 256 + threadIdx.x;
    if (h >= HID) return;
    float i_raw = wsf[1024 + h];
    float o_raw = wsf[2048 + h];
    float g_raw = wsf[3072 + h];
    float expi  = __expf(1.0f / (1.0f + __expf(-i_raw)));  // exp(sigmoid(i))
    float gt    = tanhf(g_raw);
    float os    = 1.0f / (1.0f + __expf(-o_raw));
    float Z     = expi + wsf[4096 + h];
    float c1    = (expi * gt + wsf[5120 + h]) / Z;
    out[h]        = os * tanhf(c1);   // h_1
    out[1024 + h] = c1;               // c_1
}

// ---------------------------------------------------------------------------
extern "C" void kernel_launch(void* const* d_in, const int* in_sizes, int n_in,
                              void* d_out, int out_size, void* d_ws, size_t ws_size,
                              hipStream_t stream) {
    const float* input_ = (const float*)d_in[0];
    const float* c_inp  = (const float*)d_in[1];
    const float* h0     = (const float*)d_in[2];
    /* d_in[3] = c_0 (unused by reference) */
    const float* wih    = (const float*)d_in[4];
    const float* whh    = (const float*)d_in[5];
    const float* awih   = (const float*)d_in[6];
    const float* awhh   = (const float*)d_in[7];
    const float* bias   = (const float*)d_in[8];
    const float* abias  = (const float*)d_in[9];

    float*  wsf = (float*)d_ws;
    __bf16* BT  = (__bf16*)((char*)d_ws + (size_t)6144 * sizeof(float));
    float*  out = (float*)d_out;

    zero_kernel<<<24, 256, 0, stream>>>(wsf, 6144);
    gates_kernel<<<64, 256, 0, stream>>>(input_, h0, wih, whh, awih, bias, abias, wsf);
    bt_kernel<<<4096, 256, 0, stream>>>(awhh, BT);
    lstm_child_kernel<<<CNUM / 16, 256, 0, stream>>>(
        c_inp, BT, /*aw=*/wsf, /*S=*/wsf + 4096, /*P=*/wsf + 5120);
    finalize_kernel<<<4, 256, 0, stream>>>(wsf, out);
}
